// GCNWalkerAgent_45071386804474
// MI455X (gfx1250) — matrix operand order repaired and checked by reading, not compiled, not run on hardware
//
#include <hip/hip_runtime.h>

// CDNA5 / gfx1250, wave32. Fused GCN blocks: WMMA f32<=f16 16x16x32,
// TDM (tensor_load_to_lds) weight staging, LDS-parked activation tiles.

typedef __attribute__((ext_vector_type(16))) _Float16 v16h;
typedef __attribute__((ext_vector_type(8)))  _Float16 v8h;
typedef __attribute__((ext_vector_type(8)))  float    v8f;
typedef unsigned int u32x4 __attribute__((ext_vector_type(4)));
typedef int          i32x8 __attribute__((ext_vector_type(8)));
typedef int          i32x4 __attribute__((ext_vector_type(4)));

#define DEG 9        // edge_src = repeat(arange(N), K+1) => deg==9, w = 1/9
#define YROW 264     // padded LDS row stride (f16) -> conflict-free ds_load_b128

// ---------------------------------------------------------------------------
// Tensor Data Mover: 1-D DMA of nbytes (mult of 4, <= 64KB) global -> LDS.
// D# per ISA 8.3/8.4: group0 {count=1, lds_addr, global_addr, type=2};
// group1 {data_size=4B, tensor_dim0=tile_dim0=ndwords, tensor_dim1=1}.
// Tracked by TENSORcnt; EXEC ignored (issue once per block from wave 0).
// ---------------------------------------------------------------------------
__device__ __forceinline__ void tdm_load_1d(void* lds_dst, const void* gsrc,
                                            unsigned nbytes) {
  unsigned lds_addr = (unsigned)(size_t)lds_dst;  // flat LDS-aperture addr[31:0]
  unsigned long long ga = (unsigned long long)(size_t)gsrc;
  unsigned ndw = nbytes >> 2;
  u32x4 g0 = { 1u, lds_addr,
               (unsigned)(ga & 0xffffffffu),
               (unsigned)((ga >> 32) & 0x1ffffffu) | (2u << 30) };
  i32x8 g1 = { (int)(2u << 16),
               (int)((ndw & 0xffffu) << 16),
               (int)(((ndw >> 16) & 0xffffu) | (1u << 16)),
               (int)((ndw & 0xffffu) << 16),
               0,
               (int)ndw,
               0, 0 };
  i32x4 z4 = {0, 0, 0, 0};
#if __clang_major__ >= 23
  i32x8 z8 = {0, 0, 0, 0, 0, 0, 0, 0};
  __builtin_amdgcn_tensor_load_to_lds(g0, g1, z4, z4, z8, 0);
#else
  __builtin_amdgcn_tensor_load_to_lds(g0, g1, z4, z4, 0);
#endif
}

// ---------------------------------------------------------------------------
// Repack row-major f32 weight [Kd x Nc] -> WMMA B-fragment layout (f16).
// Slot j of lane l holds B[K = (j&7) + 8*(l>>4) + 16*(j>>3)][N = l&15]; tile = n*numK + kc,
// 512 f16 per tile, lane-major (32B contiguous per lane).
// ---------------------------------------------------------------------------
__global__ void repack_b_kernel(const float* __restrict__ W, _Float16* __restrict__ Bp,
                                int Kd, int Nc) {
  int idx = blockIdx.x * blockDim.x + threadIdx.x;
  if (idx >= Kd * Nc) return;
  int tile = idx >> 9;
  int r    = idx & 511;
  int lane = r >> 4;
  int j    = r & 15;
  int numK = Kd >> 5;
  int n    = tile / numK;
  int kc   = tile - n * numK;
  int krow = kc * 32 + (j & 7) + ((lane >> 4) << 3) + ((j >> 3) << 4);
  int col  = n * 16 + (lane & 15);
  Bp[idx] = (_Float16)W[(size_t)krow * Nc + col];
}

__device__ __forceinline__ v16h make_a_frag(const v8h lo, const v8h hi) {
  return __builtin_shufflevector(lo, hi, 0,1,2,3,4,5,6,7,8,9,10,11,12,13,14,15);
}

// ELU with hardware exp (v_exp_f32): result is quantized to f16 downstream,
// so exp(x)-1 cancellation (~2^-24 abs) is far below output precision.
__device__ __forceinline__ float elu_f(float x) {
  return x > 0.f ? x : (__expf(x) - 1.0f);
}

// ---------------------------------------------------------------------------
// Fused conv block: per wave (16 rows):
//   A) agg = (1/9) sum of 9 neighbor rows of h_in -> A fragments (f16)
//   B) Y = elu(agg @ W1 + b1)  (WMMA, B1 in LDS)  -> LDS tile [16 x 256]
//   C) out = Y @ W2 + b2 + h_in (residual), row L2-normalize -> h_out
// h ping-pongs between buffers (aggregation of other blocks reads h_in).
// Requires M % 16 == 0 (N=100000 ok).
// ---------------------------------------------------------------------------
__global__ __launch_bounds__(256) void fused_conv(
    const float* __restrict__ h_in, float* __restrict__ h_out,
    const int* __restrict__ edge_dst,
    const _Float16* __restrict__ B1p, const float* __restrict__ b1,
    const _Float16* __restrict__ B2p, const float* __restrict__ b2,
    int M) {
  extern __shared__ char smem_raw[];
  _Float16* sB1 = (_Float16*)smem_raw;                    // 64x256 packed, 32KB
  _Float16* sB2 = (_Float16*)(smem_raw + 32768);          // 256x64 packed, 32KB
  const int wave = threadIdx.x >> 5;
  const int lane = threadIdx.x & 31;
  _Float16* sY = (_Float16*)(smem_raw + 65536) + (size_t)wave * (16 * YROW);

  if (wave == 0) {
    tdm_load_1d(sB1, B1p, 64 * 256 * 2);
    tdm_load_1d(sB2, B2p, 256 * 64 * 2);
    __builtin_amdgcn_s_wait_tensorcnt(0);
  }
  __syncthreads();

  const int rt = (blockIdx.x * 8 + wave) * 16;    // wave-uniform
  if (rt >= M) return;
  const int laneRow = lane & 15;
  const int half    = lane >> 4;
  const int row     = rt + laneRow;
  const int cbase   = half << 3;                  // 0 or 8

  // ---- Phase A: on-the-fly aggregation into A fragments -------------------
  const int* ed = edge_dst + (size_t)row * DEG;
  float acc[4][8];
#pragma unroll
  for (int q = 0; q < 4; ++q)
#pragma unroll
    for (int s = 0; s < 8; ++s) acc[q][s] = 0.f;
#pragma unroll
  for (int j = 0; j < DEG; ++j) {
    const float* hr = h_in + (size_t)ed[j] * 64 + cbase;
#pragma unroll
    for (int q = 0; q < 4; ++q) {
      float4 p0 = *(const float4*)(hr + q * 16);
      float4 p1 = *(const float4*)(hr + q * 16 + 4);
      acc[q][0] += p0.x; acc[q][1] += p0.y; acc[q][2] += p0.z; acc[q][3] += p0.w;
      acc[q][4] += p1.x; acc[q][5] += p1.y; acc[q][6] += p1.z; acc[q][7] += p1.w;
    }
  }
  v16h af[2];
  const float wdeg = 1.0f / (float)DEG;
#pragma unroll
  for (int kc = 0; kc < 2; ++kc)
#pragma unroll
    for (int s = 0; s < 16; ++s)
      af[kc][s] = (_Float16)(acc[kc * 2 + (s >> 3)][s & 7] * wdeg);

  // ---- Phase B: GEMM1 + bias + ELU -> LDS Y tile (row-major, padded) ------
#pragma unroll 1
  for (int n = 0; n < 16; ++n) {
    v8f c = {};
    const _Float16* bt = sB1 + (n * 2) * 512 + lane * 16;
    v16h bf0 = *(const v16h*)bt;
    v16h bf1 = *(const v16h*)(bt + 512);
    c = __builtin_amdgcn_wmma_f32_16x16x32_f16(false, af[0], false, bf0, (short)0, c, false, false);
    c = __builtin_amdgcn_wmma_f32_16x16x32_f16(false, af[1], false, bf1, (short)0, c, false, false);
    const int col = n * 16 + laneRow;
    const float bv = b1[col];
#pragma unroll
    for (int j = 0; j < 8; ++j)
      sY[(j + (half << 3)) * YROW + col] = (_Float16)elu_f(c[j] + bv);
  }
  __asm__ volatile("" ::: "memory");   // keep LDS stores before reloads
  // per-wave LDS ops are hardware-in-order: no barrier needed (own tile only)

  // ---- Phase C: GEMM2 (K=256) + bias + residual + L2 normalize ------------
  v8f c2[4] = {};
  for (int kc = 0; kc < 8; ++kc) {
    const _Float16* yp = sY + laneRow * YROW + kc * 32 + cbase;
    v16h afy = make_a_frag(*(const v8h*)yp, *(const v8h*)(yp + 16));
#pragma unroll
    for (int n = 0; n < 4; ++n) {
      v16h bf = *(const v16h*)(sB2 + (n * 8 + kc) * 512 + lane * 16);
      c2[n] = __builtin_amdgcn_wmma_f32_16x16x32_f16(false, afy, false, bf, (short)0, c2[n], false, false);
    }
  }
#pragma unroll
  for (int n = 0; n < 4; ++n) {
    const int col = n * 16 + laneRow;
    const float bv = b2[col];
#pragma unroll
    for (int j = 0; j < 8; ++j)
      c2[n][j] = c2[n][j] + bv + h_in[(size_t)(rt + j + (half << 3)) * 64 + col];
  }
#pragma unroll
  for (int j = 0; j < 8; ++j) {
    float s = 0.f;
#pragma unroll
    for (int n = 0; n < 4; ++n) s += c2[n][j] * c2[n][j];
    s += __shfl_xor(s, 1);
    s += __shfl_xor(s, 2);
    s += __shfl_xor(s, 4);
    s += __shfl_xor(s, 8);
    const float inv = 1.0f / fmaxf(sqrtf(s), 1e-12f);
    const int grow = rt + j + (half << 3);
#pragma unroll
    for (int n = 0; n < 4; ++n)
      h_out[(size_t)grow * 64 + n * 16 + laneRow] = c2[n][j] * inv;
  }
}

// ---------------------------------------------------------------------------
// Fused MLP head: gather h[ids] -> elu(.@nW1+b1) -> @nW2+b2 -> f32 out [M,128]
// ---------------------------------------------------------------------------
__global__ __launch_bounds__(256) void fused_head(
    const float* __restrict__ h_in, const int* __restrict__ ids,
    const _Float16* __restrict__ B1p, const float* __restrict__ b1,
    const _Float16* __restrict__ B2p, const float* __restrict__ b2,
    float* __restrict__ out, int M) {
  extern __shared__ char smem_raw[];
  _Float16* sB1 = (_Float16*)smem_raw;                    // 64x256 packed, 32KB
  _Float16* sB2 = (_Float16*)(smem_raw + 32768);          // 256x128 packed, 64KB
  const int wave = threadIdx.x >> 5;
  const int lane = threadIdx.x & 31;
  _Float16* sY = (_Float16*)(smem_raw + 32768 + 65536) + (size_t)wave * (16 * YROW);

  if (wave == 0) {
    tdm_load_1d(sB1, B1p, 64 * 256 * 2);
    tdm_load_1d(sB2, B2p, 256 * 128 * 2);
    __builtin_amdgcn_s_wait_tensorcnt(0);
  }
  __syncthreads();

  const int rt = (blockIdx.x * 8 + wave) * 16;
  if (rt >= M) return;
  const int laneRow = lane & 15;
  const int half    = lane >> 4;
  const int cbase   = half << 3;

  // ---- Phase A: gather one row, build A fragments -------------------------
  const float* hr = h_in + (size_t)ids[rt + laneRow] * 64 + cbase;
  v16h af[2];
#pragma unroll
  for (int kc = 0; kc < 2; ++kc) {
#pragma unroll
    for (int q = 0; q < 2; ++q) {
      float4 p0 = *(const float4*)(hr + (kc * 2 + q) * 16);
      float4 p1 = *(const float4*)(hr + (kc * 2 + q) * 16 + 4);
      af[kc][q * 8 + 0] = (_Float16)p0.x; af[kc][q * 8 + 1] = (_Float16)p0.y;
      af[kc][q * 8 + 2] = (_Float16)p0.z; af[kc][q * 8 + 3] = (_Float16)p0.w;
      af[kc][q * 8 + 4] = (_Float16)p1.x; af[kc][q * 8 + 5] = (_Float16)p1.y;
      af[kc][q * 8 + 6] = (_Float16)p1.z; af[kc][q * 8 + 7] = (_Float16)p1.w;
    }
  }

  // ---- Phase B: layer 1 -> LDS Y tile -------------------------------------
#pragma unroll 1
  for (int n = 0; n < 16; ++n) {
    v8f c = {};
    const _Float16* bt = sB1 + (n * 2) * 512 + lane * 16;
    v16h bf0 = *(const v16h*)bt;
    v16h bf1 = *(const v16h*)(bt + 512);
    c = __builtin_amdgcn_wmma_f32_16x16x32_f16(false, af[0], false, bf0, (short)0, c, false, false);
    c = __builtin_amdgcn_wmma_f32_16x16x32_f16(false, af[1], false, bf1, (short)0, c, false, false);
    const int col = n * 16 + laneRow;
    const float bv = b1[col];
#pragma unroll
    for (int j = 0; j < 8; ++j)
      sY[(j + (half << 3)) * YROW + col] = (_Float16)elu_f(c[j] + bv);
  }
  __asm__ volatile("" ::: "memory");

  // ---- Phase C: layer 2 (K=256, Nc=128) -> f32 out ------------------------
  v16h afy[8];
#pragma unroll
  for (int kc = 0; kc < 8; ++kc) {
    const _Float16* yp = sY + laneRow * YROW + kc * 32 + cbase;
    afy[kc] = make_a_frag(*(const v8h*)yp, *(const v8h*)(yp + 16));
  }
#pragma unroll 1
  for (int n = 0; n < 8; ++n) {
    v8f c = {};
#pragma unroll
    for (int kc = 0; kc < 8; ++kc) {
      v16h bf = *(const v16h*)(sB2 + (n * 8 + kc) * 512 + lane * 16);
      c = __builtin_amdgcn_wmma_f32_16x16x32_f16(false, afy[kc], false, bf, (short)0, c, false, false);
    }
    const int col = n * 16 + laneRow;
    const float bv = b2[col];
#pragma unroll
    for (int j = 0; j < 8; ++j)
      out[(size_t)(rt + j + (half << 3)) * 128 + col] = c[j] + bv;
  }
}

// ---------------------------------------------------------------------------
extern "C" void kernel_launch(void* const* d_in, const int* in_sizes, int n_in,
                              void* d_out, int out_size, void* d_ws, size_t ws_size,
                              hipStream_t stream) {
  (void)n_in; (void)out_size; (void)ws_size;
  const float* vertices   = (const float*)d_in[0];
  const int*   edge_dst   = (const int*)d_in[2];
  const int*   vertex_ids = (const int*)d_in[3];
  const float* cW1 = (const float*)d_in[4];
  const float* cb1 = (const float*)d_in[5];
  const float* cW2 = (const float*)d_in[6];
  const float* cb2 = (const float*)d_in[7];
  const float* nW1 = (const float*)d_in[8];
  const float* nb1 = (const float*)d_in[9];
  const float* nW2 = (const float*)d_in[10];
  const float* nb2 = (const float*)d_in[11];

  const int N  = in_sizes[0] / 64;   // 100000 (multiple of 16)
  const int Bq = in_sizes[3];        // 16384  (multiple of 16)

  char* ws = (char*)d_ws;
  size_t off = 0;
  auto carve = [&](size_t bytes) -> void* {
    void* p = ws + off;
    off = (off + bytes + 255) & ~(size_t)255;
    return p;
  };
  float*    hA   = (float*)   carve((size_t)N * 64 * 4);  // ping
  float*    hB   = (float*)   carve((size_t)N * 64 * 4);  // pong
  _Float16* W1p  = (_Float16*)carve((size_t)3 * 64 * 256 * 2);
  _Float16* W2p  = (_Float16*)carve((size_t)3 * 256 * 64 * 2);
  _Float16* nW1p = (_Float16*)carve((size_t)64 * 256 * 2);
  _Float16* nW2p = (_Float16*)carve((size_t)256 * 128 * 2);

  hipMemcpyAsync(hA, vertices, (size_t)N * 64 * 4, hipMemcpyDeviceToDevice, stream);

  auto repack = [&](const float* W, _Float16* P, int Kd, int Nc) {
    int total = Kd * Nc;
    repack_b_kernel<<<(total + 255) / 256, 256, 0, stream>>>(W, P, Kd, Nc);
  };
  for (int i = 0; i < 3; ++i) {
    repack(cW1 + (size_t)i * 64 * 256, W1p + (size_t)i * 64 * 256, 64, 256);
    repack(cW2 + (size_t)i * 256 * 64, W2p + (size_t)i * 256 * 64, 256, 64);
  }
  repack(nW1, nW1p, 64, 256);
  repack(nW2, nW2p, 256, 128);

  const int gb = (N + 127) / 128;
  const unsigned convLds = 32768 + 32768 + 8 * 16 * YROW * 2;        // 130 KB
  const unsigned headLds = 32768 + 65536 + 8 * 16 * YROW * 2;        // 162 KB
  float* hin = hA;
  float* hout = hB;
  for (int i = 0; i < 3; ++i) {
    fused_conv<<<gb, 256, convLds, stream>>>(
        hin, hout, edge_dst,
        W1p + (size_t)i * 64 * 256, cb1 + (size_t)i * 256,
        W2p + (size_t)i * 256 * 64, cb2 + (size_t)i * 64, N);
    float* t = hin; hin = hout; hout = t;
  }
  // after 3 swaps, hin points at the latest hidden state
  fused_head<<<(Bq + 127) / 128, 256, headLds, stream>>>(
      hin, vertex_ids, nW1p, nb1, nW2p, nb2, (float*)d_out, Bq);
}
